// CausalSelfAttentionKVCache_54047868453462
// MI455X (gfx1250) — matrix operand order repaired and checked by reading, not compiled, not run on hardware
//
#include <hip/hip_runtime.h>

// ---------------- problem constants ----------------
#define BATCH   8
#define SEQ     16
#define NH      16
#define HD      128
#define DMODEL  2048
#define D3      (3 * DMODEL)       // 6144
#define TCACHE  4096
#define TTOT    4112               // 4096 + 16

// d_out layout (floats): out | kh | vh
#define KH_OFF  ((size_t)(BATCH * SEQ * DMODEL))                    // 262144
#define VH_OFF  ((size_t)(KH_OFF + (size_t)BATCH * NH * TTOT * HD)) // 67633152

// workspace layout (bytes)
#define XB_OFF   ((size_t)0)                               // x bf16 [128,2048]
#define WTA_OFF  ((size_t)(XB_OFF + 524288))               // W_attn^T bf16 [6144,2048]
#define WTP_OFF  ((size_t)(WTA_OFF + 25165824))            // W_proj^T bf16 [2048,2048]
#define QB_OFF   ((size_t)(WTP_OFF + 8388608))             // q bf16 [B,H,S,HD]
#define CTX_OFF  ((size_t)(QB_OFF + 524288))               // ctx bf16 [128,2048]
#define PART_OFF ((size_t)(CTX_OFF + 524288))              // partials f32 [512][2080]
#define PART_STRIDE 2080                                   // 2048 acc + 16 m + 16 l

typedef __attribute__((ext_vector_type(16))) __bf16 v16bf;
typedef __attribute__((ext_vector_type(8)))  __bf16 v8bf;
typedef __attribute__((ext_vector_type(8)))  float  v8f;
typedef __attribute__((ext_vector_type(4)))  unsigned int u32x4;
typedef __attribute__((ext_vector_type(8)))  int i32x8;
typedef __attribute__((ext_vector_type(4)))  int i32x4;

// ---------------- WMMA fragment helpers ----------------
// A-matrix fragment (16x32 bf16): lane (m=l&15, hi=l>>4) holds
// K = [hi*16, hi*16+16) of row m, contiguous (matches B fragment)
__device__ __forceinline__ v16bf load_fragA(const __bf16* rowptr, int hi) {
  return *(const v16bf*)(rowptr + hi * 16);
}
// B-matrix fragment (32x16 bf16): lane (n=l&15, hi=l>>4) holds K=[hi*16,hi*16+16)
// of column n, contiguous (B^T row-major) -> one aligned 32B load
__device__ __forceinline__ v16bf load_fragB(const __bf16* rowptr, int hi) {
  return *(const v16bf*)(rowptr + hi * 16);
}
__device__ __forceinline__ v8f wmma_bf16(v16bf a, v16bf b, v8f c) {
  return __builtin_amdgcn_wmma_f32_16x16x32_bf16(false, a, false, b,
                                                 (short)0, c, false, false);
}

// ---------------- TDM descriptor helpers (ISA 08_async_tensor §8) ----------------
// 2-D fp32 tile: tile_dim0 = 128 (contiguous), tile_dim1 = rows, stride0 = 128
__device__ __forceinline__ u32x4 tdm_g0(unsigned lds, const void* g) {
  unsigned long long ga = (unsigned long long)g;
  u32x4 r;
  r[0] = 1u;                                   // count=1 (valid user descriptor)
  r[1] = lds;                                  // lds_addr (bytes)
  r[2] = (unsigned)ga;                         // global_addr[31:0]
  r[3] = (unsigned)(ga >> 32) | (2u << 30);    // global_addr[56:32] | type=2
  return r;
}
__device__ __forceinline__ i32x8 tdm_g1(int rows) {
  i32x8 r;
  r[0] = 0x20000;       // data_size = 2 (4 bytes); no mask/pad/iterate
  r[1] = HD << 16;      // tensor_dim0[15:0] = 128
  r[2] = rows << 16;    // tensor_dim0[31:16]=0 | tensor_dim1[15:0] = rows
  r[3] = HD << 16;      // tensor_dim1[31:16]=0 | tile_dim0 = 128
  r[4] = rows;          // tile_dim1 = rows, tile_dim2 = 0
  r[5] = HD;            // tensor_dim0_stride[31:0] = 128
  r[6] = 0;
  r[7] = 0;
  return r;
}
__device__ __forceinline__ void tdm_load(unsigned lds, const void* g, int rows) {
  const i32x4 z4 = {0, 0, 0, 0};
#if defined(__clang_major__) && __clang_major__ >= 23
  const i32x8 z8 = {0, 0, 0, 0, 0, 0, 0, 0};
  __builtin_amdgcn_tensor_load_to_lds(tdm_g0(lds, g), tdm_g1(rows), z4, z4, z8, 0);
#else
  __builtin_amdgcn_tensor_load_to_lds(tdm_g0(lds, g), tdm_g1(rows), z4, z4, 0);
#endif
}
__device__ __forceinline__ void tdm_store(unsigned lds, void* g, int rows) {
  const i32x4 z4 = {0, 0, 0, 0};
#if defined(__clang_major__) && __clang_major__ >= 23
  const i32x8 z8 = {0, 0, 0, 0, 0, 0, 0, 0};
  __builtin_amdgcn_tensor_store_from_lds(tdm_g0(lds, g), tdm_g1(rows), z4, z4, z8, 0);
#else
  __builtin_amdgcn_tensor_store_from_lds(tdm_g0(lds, g), tdm_g1(rows), z4, z4, 0);
#endif
}

// ---------------- prep kernels ----------------
__global__ __launch_bounds__(256) void convert_x_kernel(const float* __restrict__ x,
                                                        __bf16* __restrict__ xb) {
  int i = blockIdx.x * 256 + threadIdx.x;
  xb[i] = (__bf16)x[i];
}

// W [K,N] f32 -> Wt [N,K] bf16, tiled 32x32
__global__ __launch_bounds__(256) void transpose_bf16_kernel(const float* __restrict__ W,
                                                             __bf16* __restrict__ Wt,
                                                             int K, int N) {
  __shared__ float tile[32][33];
  int k0 = blockIdx.x * 32, n0 = blockIdx.y * 32;
  int tx = threadIdx.x, ty = threadIdx.y;
#pragma unroll
  for (int i = 0; i < 32; i += 8)
    tile[ty + i][tx] = W[(size_t)(k0 + ty + i) * N + n0 + tx];
  __syncthreads();
#pragma unroll
  for (int i = 0; i < 32; i += 8)
    Wt[(size_t)(n0 + ty + i) * K + k0 + tx] = (__bf16)tile[tx][ty + i];
}

// ---------------- GEMM core: one wave computes C[64, 16] panel (4 m-tiles) ------
__device__ __forceinline__ void gemm_core4(const __bf16* __restrict__ A,  // [128,K]
                                           const __bf16* __restrict__ Bt, // [N,K]
                                           int K, int n0, int mt0, int lane,
                                           v8f acc[4]) {
  const int nl = lane & 15, hi = lane >> 4;
  const __bf16* brow = Bt + (size_t)(n0 + nl) * K;
  for (int k0 = 0; k0 < K; k0 += 32) {
    __builtin_prefetch(brow + k0 + 512, 0, 1);   // global_prefetch_b8
    v16bf bf = load_fragB(brow + k0, hi);
#pragma unroll
    for (int mt = 0; mt < 4; ++mt) {
      const __bf16* arow = A + (size_t)((mt0 + mt) * 16 + nl) * K + k0;
      acc[mt] = wmma_bf16(load_fragA(arow, hi), bf, acc[mt]);
    }
  }
}

// QKV GEMM + scatter: cols [0,2048)->q ws, [2048,4096)->kh tail, [4096,6144)->vh tail
__global__ __launch_bounds__(64) void gemm_qkv_kernel(const __bf16* __restrict__ xb,
                                                      const __bf16* __restrict__ wta,
                                                      const float* __restrict__ bias,
                                                      __bf16* __restrict__ qout,
                                                      float* __restrict__ out) {
  const int lane = threadIdx.x & 31, wid = threadIdx.x >> 5;
  const int nl = lane & 15, hi = lane >> 4;
  const int n0 = blockIdx.x * 16, mt0 = wid * 4;
  v8f acc[4];
#pragma unroll
  for (int mt = 0; mt < 4; ++mt)
#pragma unroll
    for (int e = 0; e < 8; ++e) acc[mt][e] = 0.f;

  gemm_core4(xb, wta, DMODEL, n0, mt0, lane, acc);

  const int col = n0 + nl;
  const float bv = bias[col];
#pragma unroll
  for (int mt = 0; mt < 4; ++mt) {
#pragma unroll
    for (int e = 0; e < 8; ++e) {
      const int row = (mt0 + mt) * 16 + e + hi * 8;   // 0..127
      const int b = row >> 4, s = row & 15;
      const float val = acc[mt][e] + bv;
      if (col < DMODEL) {
        const int h = col >> 7, d = col & 127;
        qout[(((size_t)(b * NH + h) * SEQ) + s) * HD + d] = (__bf16)val;
      } else if (col < 2 * DMODEL) {
        const int c = col - DMODEL, h = c >> 7, d = c & 127;
        out[KH_OFF + ((size_t)(b * NH + h) * TTOT + TCACHE + s) * HD + d] = val;
      } else {
        const int c = col - 2 * DMODEL, h = c >> 7, d = c & 127;
        out[VH_OFF + ((size_t)(b * NH + h) * TTOT + TCACHE + s) * HD + d] = val;
      }
    }
  }
}

// Projection GEMM: out[row, col] into d_out[0 : 128*2048)
__global__ __launch_bounds__(64) void gemm_proj_kernel(const __bf16* __restrict__ ctx,
                                                       const __bf16* __restrict__ wtp,
                                                       const float* __restrict__ bias,
                                                       float* __restrict__ out) {
  const int lane = threadIdx.x & 31, wid = threadIdx.x >> 5;
  const int nl = lane & 15, hi = lane >> 4;
  const int n0 = blockIdx.x * 16, mt0 = wid * 4;
  v8f acc[4];
#pragma unroll
  for (int mt = 0; mt < 4; ++mt)
#pragma unroll
    for (int e = 0; e < 8; ++e) acc[mt][e] = 0.f;

  gemm_core4(ctx, wtp, DMODEL, n0, mt0, lane, acc);

  const int col = n0 + nl;
  const float bv = bias[col];
#pragma unroll
  for (int mt = 0; mt < 4; ++mt)
#pragma unroll
    for (int e = 0; e < 8; ++e) {
      const int row = (mt0 + mt) * 16 + e + hi * 8;
      out[(size_t)row * DMODEL + col] = acc[mt][e] + bv;
    }
}

// ---------------- fused flash attention + cache copy (partitioned) ----------------
// grid = B*H*4 workgroups (4 T-partitions per head); 4 waves/WG.
// Global chunk c (32 keys, c=0..128; 128 = 16-row tail) handled by
// part p = c&3, wave w = (c>>2)&3  (c ≡ p+4w mod 16).
// Per-wave LDS region (52288 B):
//   kf32 16384 | vf32 16384 | kt 8192 | vt 8192 | st 2048 | pt 1024 | sc 64
#define KF_O  0
#define VF_O  16384
#define KT_O  32768
#define VT_O  40960
#define ST_O  49152
#define PT_O  51200
#define SC_O  52224
#define PER_WAVE 52288
#define BIG_NEG  (-1e30f)

__global__ __launch_bounds__(128) void attn_part_kernel(const float* __restrict__ kc,
                                                        const float* __restrict__ vc,
                                                        const __bf16* __restrict__ qb,
                                                        float* __restrict__ out,
                                                        float* __restrict__ part) {
  extern __shared__ char smem[];
  const int tid = threadIdx.x;
  const int wid = tid >> 5, lane = tid & 31;
  const int nl = lane & 15, hi = lane >> 4;
  const int wgid = blockIdx.x;
  const int bh = wgid >> 2, p = wgid & 3;

  char* wbase = smem + (size_t)wid * PER_WAVE;
  float*  kf = (float*) (wbase + KF_O);    // [32 keys][128] fp32 (TDM landing)
  float*  vf = (float*) (wbase + VF_O);
  __bf16* kt = (__bf16*)(wbase + KT_O);    // [32 keys][128] bf16
  __bf16* vt = (__bf16*)(wbase + VT_O);    // [128 hd][32 keys] bf16 (transposed)
  float*  st = (float*) (wbase + ST_O);    // [16 rows][32] scores
  __bf16* pt = (__bf16*)(wbase + PT_O);    // [16 rows][32] probs
  float*  sc = (float*) (wbase + SC_O);    // [16] rescale
  const unsigned kf_lds = (unsigned)(uintptr_t)(wbase + KF_O);
  const unsigned vf_lds = (unsigned)(uintptr_t)(wbase + VF_O);

  const float* kcb = kc + (size_t)bh * TCACHE * HD;
  const float* vcb = vc + (size_t)bh * TCACHE * HD;
  float* okh = out + KH_OFF + (size_t)bh * TTOT * HD;
  float* ovh = out + VH_OFF + (size_t)bh * TTOT * HD;
  const __bf16* qp = qb + (size_t)bh * SEQ * HD;

  v16bf qf[4];
#pragma unroll
  for (int ks = 0; ks < 4; ++ks) qf[ks] = load_fragA(qp + nl * HD + ks * 32, hi);

  v8f cacc[8];
#pragma unroll
  for (int nt = 0; nt < 8; ++nt)
#pragma unroll
    for (int e = 0; e < 8; ++e) cacc[nt][e] = 0.f;

  float run_m = BIG_NEG, run_l = 0.f;
  const float rscale = 0.08838834764831845f;  // 1/sqrt(128)

  for (int c = p + 4 * wid; c <= 128; c += 16) {
    const bool tail = (c == 128);
    const int nrows = tail ? 16 : 32;
    const float* ksrc = tail ? (okh + (size_t)TCACHE * HD) : (kcb + (size_t)c * 32 * HD);
    const float* vsrc = tail ? (ovh + (size_t)TCACHE * HD) : (vcb + (size_t)c * 32 * HD);

    // ---- TDM: DMA fp32 tiles global -> LDS (one instruction each)
    tdm_load(kf_lds, ksrc, nrows);
    tdm_load(vf_lds, vsrc, nrows);
    __builtin_amdgcn_s_wait_tensorcnt((short)0);

    // ---- fused cache->output copy, LDS -> global via TDM (background;
    //      same-wave tensor ops stay in order, so next chunk's load is safe)
    if (!tail) {
      tdm_store(kf_lds, okh + (size_t)c * 32 * HD, 32);
      tdm_store(vf_lds, ovh + (size_t)c * 32 * HD, 32);
    }

    // ---- convert LDS fp32 -> bf16 tiles (kt row-major, vt transposed)
#pragma unroll 4
    for (int r = 0; r < 32; ++r) {
      __bf16* kd = kt + r * HD + lane * 4;
      if (r < nrows) {
        const float4 kv = *(const float4*)(kf + r * HD + lane * 4);
        kd[0] = (__bf16)kv.x; kd[1] = (__bf16)kv.y; kd[2] = (__bf16)kv.z; kd[3] = (__bf16)kv.w;
        const float4 vv = *(const float4*)(vf + r * HD + lane * 4);
        vt[(lane * 4 + 0) * 32 + r] = (__bf16)vv.x;
        vt[(lane * 4 + 1) * 32 + r] = (__bf16)vv.y;
        vt[(lane * 4 + 2) * 32 + r] = (__bf16)vv.z;
        vt[(lane * 4 + 3) * 32 + r] = (__bf16)vv.w;
      } else {
        const __bf16 z = (__bf16)0.f;
        kd[0] = z; kd[1] = z; kd[2] = z; kd[3] = z;
        vt[(lane * 4 + 0) * 32 + r] = z; vt[(lane * 4 + 1) * 32 + r] = z;
        vt[(lane * 4 + 2) * 32 + r] = z; vt[(lane * 4 + 3) * 32 + r] = z;
      }
    }
    asm volatile("s_wait_dscnt 0" ::: "memory");

    // ---- scores: S[16x32] = Q[16x128] * Kchunk^T (two 16-key tiles, K=128)
    v8f sf0, sf1;
#pragma unroll
    for (int e = 0; e < 8; ++e) { sf0[e] = 0.f; sf1[e] = 0.f; }
#pragma unroll
    for (int ks = 0; ks < 4; ++ks) {
      sf0 = wmma_bf16(qf[ks], load_fragB(kt + (size_t)nl * HD + ks * 32, hi), sf0);
      sf1 = wmma_bf16(qf[ks], load_fragB(kt + (size_t)(16 + nl) * HD + ks * 32, hi), sf1);
    }
#pragma unroll
    for (int e = 0; e < 8; ++e) {
      st[(e + hi * 8) * 32 + nl]      = sf0[e];
      st[(e + hi * 8) * 32 + 16 + nl] = sf1[e];
    }
    asm volatile("s_wait_dscnt 0" ::: "memory");

    // ---- online softmax: lane i owns row i (lanes 0..15)
    if (lane < 16) {
      const int i = lane;
      float srow[32];
      float mx = BIG_NEG;
#pragma unroll
      for (int j = 0; j < 32; ++j) {
        float s = st[i * 32 + j] * rscale;
        const bool ok = (!tail) || ((j < 16) && (j <= i));
        s = ok ? s : BIG_NEG;
        srow[j] = s;
        mx = fmaxf(mx, s);
      }
      const float nm = fmaxf(run_m, mx);
      const float rs = __expf(run_m - nm);
      float sum = 0.f;
#pragma unroll
      for (int j = 0; j < 32; ++j) {
        const float pv = __expf(srow[j] - nm);
        sum += pv;
        pt[i * 32 + j] = (__bf16)pv;
      }
      run_l = run_l * rs + sum;
      run_m = nm;
      sc[i] = rs;
    }
    asm volatile("s_wait_dscnt 0" ::: "memory");

    // ---- rescale accumulators
#pragma unroll
    for (int e = 0; e < 8; ++e) {
      const float rs = sc[e + hi * 8];
#pragma unroll
      for (int nt = 0; nt < 8; ++nt) cacc[nt][e] *= rs;
    }

    // ---- ctx += P[16x32] * V[32x128]
    const v16bf pf = load_fragA(pt + nl * 32, hi);
#pragma unroll
    for (int nt = 0; nt < 8; ++nt)
      cacc[nt] = wmma_bf16(pf, load_fragB(vt + (size_t)(nt * 16 + nl) * 32, hi), cacc[nt]);
  }

  // drain outstanding copy stores before reusing kf region for combine
  __builtin_amdgcn_s_wait_tensorcnt((short)0);

  // ---- cross-wave combine (reuse kf region), write WG partial to workspace
  float* accw = (float*)(wbase);            // [16][128]
  float* mw   = (float*)(wbase + 8192);     // [16]
  float* lw   = (float*)(wbase + 8256);     // [16]
#pragma unroll
  for (int nt = 0; nt < 8; ++nt)
#pragma unroll
    for (int e = 0; e < 8; ++e)
      accw[(e + hi * 8) * HD + nt * 16 + nl] = cacc[nt][e];
  if (lane < 16) { mw[lane] = run_m; lw[lane] = run_l; }
  __syncthreads();

  float* pb = part + (size_t)wgid * PART_STRIDE;
  for (int idx = tid; idx < SEQ * HD; idx += 128) {
    const int row = idx >> 7, d = idx & 127;
    float M = BIG_NEG;
#pragma unroll
    for (int w = 0; w < 4; ++w)
      M = fmaxf(M, ((float*)(smem + (size_t)w * PER_WAVE + 8192))[row]);
    float num = 0.f, den = 0.f;
#pragma unroll
    for (int w = 0; w < 4; ++w) {
      char* wb = smem + (size_t)w * PER_WAVE;
      const float f = __expf(((float*)(wb + 8192))[row] - M);
      num += f * ((float*)wb)[row * HD + d];
      den += f * ((float*)(wb + 8256))[row];
    }
    pb[row * HD + d] = num;
    if (d == 0) { pb[2048 + row] = M; pb[2064 + row] = den; }
  }
}

// ---------------- final cross-partition combine -> ctx bf16 ----------------
__global__ __launch_bounds__(128) void attn_combine_kernel(const float* __restrict__ part,
                                                           __bf16* __restrict__ ctx) {
  const int bh = blockIdx.x, b = bh >> 4, h = bh & 15;
  for (int idx = threadIdx.x; idx < SEQ * HD; idx += 128) {
    const int row = idx >> 7, d = idx & 127;
    float Mg = BIG_NEG;
#pragma unroll
    for (int p = 0; p < 4; ++p)
      Mg = fmaxf(Mg, part[(size_t)(bh * 4 + p) * PART_STRIDE + 2048 + row]);
    float num = 0.f, den = 0.f;
#pragma unroll
    for (int p = 0; p < 4; ++p) {
      const float* pb = part + (size_t)(bh * 4 + p) * PART_STRIDE;
      const float f = __expf(pb[2048 + row] - Mg);
      num += f * pb[row * HD + d];
      den += f * pb[2064 + row];
    }
    ctx[((size_t)(b * SEQ + row)) * DMODEL + h * HD + d] = (__bf16)(num / den);
  }
}

// ---------------- launcher ----------------
extern "C" void kernel_launch(void* const* d_in, const int* in_sizes, int n_in,
                              void* d_out, int out_size, void* d_ws, size_t ws_size,
                              hipStream_t stream) {
  (void)in_sizes; (void)n_in; (void)out_size; (void)ws_size;
  const float* x  = (const float*)d_in[0];
  const float* kc = (const float*)d_in[1];
  const float* vc = (const float*)d_in[2];
  const float* Wa = (const float*)d_in[3];
  const float* ba = (const float*)d_in[4];
  const float* Wp = (const float*)d_in[5];
  const float* bp = (const float*)d_in[6];
  float* out = (float*)d_out;
  char* ws = (char*)d_ws;

  __bf16* xb   = (__bf16*)(ws + XB_OFF);
  __bf16* wta  = (__bf16*)(ws + WTA_OFF);
  __bf16* wtp  = (__bf16*)(ws + WTP_OFF);
  __bf16* qb   = (__bf16*)(ws + QB_OFF);
  __bf16* ctx  = (__bf16*)(ws + CTX_OFF);
  float*  part = (float*)(ws + PART_OFF);

  // 1. x -> bf16
  convert_x_kernel<<<1024, 256, 0, stream>>>(x, xb);

  // 2. weight transposes (f32 [K,N] -> bf16 [N,K])
  transpose_bf16_kernel<<<dim3(DMODEL / 32, D3 / 32), dim3(32, 8), 0, stream>>>(Wa, wta, DMODEL, D3);
  transpose_bf16_kernel<<<dim3(DMODEL / 32, DMODEL / 32), dim3(32, 8), 0, stream>>>(Wp, wtp, DMODEL, DMODEL);

  // 3. QKV GEMM + scatter (q -> ws, new k/v -> output tails)
  gemm_qkv_kernel<<<D3 / 16, 64, 0, stream>>>(xb, wta, ba, qb, out);

  // 4. fused flash attention + TDM cache copy, 4 partitions per (b,h)
  attn_part_kernel<<<BATCH * NH * 4, 128, 4 * PER_WAVE, stream>>>(kc, vc, qb, out, part);

  // 5. combine partials -> ctx
  attn_combine_kernel<<<BATCH * NH, 128, 0, stream>>>(part, ctx);

  // 6. output projection
  gemm_proj_kernel<<<DMODEL / 16, 64, 0, stream>>>(ctx, wtp, bp, out);
}
